// LongformerClassifier_16827681865844
// MI455X (gfx1250) — matrix operand order repaired and checked by reading, not compiled, not run on hardware
//
#include <hip/hip_runtime.h>

// ---------------------------------------------------------------------------
// MI455X (gfx1250) Longformer classifier.
// - All GEMM-class work through V_WMMA_F32_16X16X32_BF16 (wave32, f32 acc).
// - Weights pre-transposed+converted to bf16 once per layer ([N][K], K-major);
//   activations keep a bf16 shadow so GEMM A/B staging is a raw byte copy.
// - LDS staging uses GLOBAL_LOAD_ASYNC_TO_LDS_B128 (ASYNCcnt) so the vector
//   pipes stay free for WMMA; fenced with s_wait_asynccnt + barrier.
// Roofline: ~2.2 TFLOP total, weights ~18MB bf16/layer (L2-resident at 192MB),
// activations ~25MB/tensor vs 23.3 TB/s HBM -> matrix-core bound by design.
// ---------------------------------------------------------------------------

typedef __attribute__((ext_vector_type(16))) __bf16 v16bf;
typedef __attribute__((ext_vector_type(8)))  __bf16 v8bf;
typedef __attribute__((ext_vector_type(8)))  float  v8f;

#define DM    768
#define FFDIM 3072
#define S_TOT 4096
#define NHEAD 12

#ifndef USE_ASYNC_LDS
#define USE_ASYNC_LDS 1
#endif

__device__ __forceinline__ float gelu_f(float x) {
  float x3 = x * x * x;
  return 0.5f * x * (1.f + tanhf(0.79788456f * (x + 0.044715f * x3)));
}

__device__ __forceinline__ v8f wmma_bf16(v16bf a, v16bf b, v8f c) {
  return __builtin_amdgcn_wmma_f32_16x16x32_bf16(
      /*neg_a=*/false, a, /*neg_b=*/false, b,
      /*c_mod=*/(short)0, c, /*reuse_a=*/false, /*reuse_b=*/false);
}

// Build a 16-element bf16 fragment from two 16-byte LDS loads.
__device__ __forceinline__ v16bf ld_frag(const __bf16* p0, const __bf16* p1) {
  union { v16bf v; v8bf h[2]; } u;
  u.h[0] = *(const v8bf*)p0;
  u.h[1] = *(const v8bf*)p1;
  return u.v;
}

// 16B global -> LDS copy. Async (TDM-free DMA path, ASYNCcnt) when enabled.
__device__ __forceinline__ void cp_b128(void* lds_dst, const void* gsrc) {
#if USE_ASYNC_LDS
  unsigned lds_off = (unsigned)(size_t)lds_dst;  // LDS aperture: low 32 bits
  asm volatile("global_load_async_to_lds_b128 %0, %1, off"
               :: "v"(lds_off), "v"(gsrc) : "memory");
#else
  *(v8bf*)lds_dst = *(const v8bf*)gsrc;
#endif
}
__device__ __forceinline__ void cp_fence() {
#if USE_ASYNC_LDS
  asm volatile("s_wait_asynccnt 0x0" ::: "memory");
#endif
}

__device__ __forceinline__ float rowmax16(float v) {
  v = fmaxf(v, __shfl_xor(v, 1, 16));
  v = fmaxf(v, __shfl_xor(v, 2, 16));
  v = fmaxf(v, __shfl_xor(v, 4, 16));
  v = fmaxf(v, __shfl_xor(v, 8, 16));
  return v;
}
__device__ __forceinline__ float rowsum16(float v) {
  v += __shfl_xor(v, 1, 16);
  v += __shfl_xor(v, 2, 16);
  v += __shfl_xor(v, 4, 16);
  v += __shfl_xor(v, 8, 16);
  return v;
}

// ---------------------------------------------------------------------------
// Weight prep: W[K][N] f32  ->  Wt[N][K] bf16 (K-contiguous for B fragments).
// 32x32 LDS-tiled transpose, block (32,8).
// ---------------------------------------------------------------------------
__global__ __launch_bounds__(256) void transpose_bf16_kernel(
    const float* __restrict__ W, __bf16* __restrict__ Wt, int K, int N) {
  __shared__ float tile[32][33];
  const int tx = threadIdx.x, ty = threadIdx.y;
  const int k0 = blockIdx.y * 32, n0 = blockIdx.x * 32;
#pragma unroll
  for (int i = 0; i < 4; ++i) {
    int r = ty + i * 8;
    tile[r][tx] = W[(size_t)(k0 + r) * N + n0 + tx];
  }
  __syncthreads();
#pragma unroll
  for (int i = 0; i < 4; ++i) {
    int r = ty + i * 8;
    Wt[(size_t)(n0 + r) * K + k0 + tx] = (__bf16)tile[tx][r];
  }
}

// ---------------------------------------------------------------------------
// C = act(A[M][K]bf16 x Bt[N][K]bf16^T + bias), optional bf16 shadow Cb.
// Block tile 128x128, BK=64 (2 WMMA k-steps), 256 thr = 8 wave32, wave 32x64.
// 16 WMMAs per staged tile; staging = 8 async b128 copies per thread.
// ---------------------------------------------------------------------------
template <int GELU_F>
__global__ __launch_bounds__(256) void gemm_kernel(
    const __bf16* __restrict__ A, const __bf16* __restrict__ Bt,
    const float* __restrict__ bias, float* __restrict__ C,
    __bf16* __restrict__ Cb, int M, int N, int K) {
  __shared__ __align__(16) __bf16 lA[128 * 64]; // [m][k]
  __shared__ __align__(16) __bf16 lB[128 * 64]; // [n][k]

  const int t = threadIdx.x;
  const int lane = t & 31;
  const int wave = t >> 5;
  const int lane15 = lane & 15;
  const int hx = lane >> 4;
  const int wm = wave >> 1;   // 0..3 (M)
  const int wn = wave & 1;    // 0..1 (N)
  const int m0 = blockIdx.y * 128;
  const int n0 = blockIdx.x * 128;

  v8f acc[2][4] = {};

  for (int k0 = 0; k0 < K; k0 += 64) {
    // ---- stage A and Bt tiles: raw bf16 copies, 16KB each ----
#pragma unroll
    for (int i = 0; i < 4; ++i) {
      int idx = t + i * 256;      // 0..1023
      int r = idx >> 3, ch = idx & 7;
      cp_b128(lA + r * 64 + ch * 8, A  + (size_t)(m0 + r) * K + k0 + ch * 8);
      cp_b128(lB + r * 64 + ch * 8, Bt + (size_t)(n0 + r) * K + k0 + ch * 8);
    }
    cp_fence();
    __syncthreads();

#pragma unroll
    for (int ks = 0; ks < 2; ++ks) {
      const int kb = ks * 32;
      v16bf af2[2], bf4[4];
#pragma unroll
      for (int mi = 0; mi < 2; ++mi) {
        int r = wm * 32 + mi * 16 + lane15;
        af2[mi] = ld_frag(lA + r * 64 + kb + hx * 8,
                          lA + r * 64 + kb + 16 + hx * 8);
      }
#pragma unroll
      for (int nj = 0; nj < 4; ++nj) {
        int cc = wn * 64 + nj * 16 + lane15;
        bf4[nj] = ld_frag(lB + cc * 64 + kb + hx * 8,
                          lB + cc * 64 + kb + 16 + hx * 8);
      }
#pragma unroll
      for (int mi = 0; mi < 2; ++mi)
#pragma unroll
        for (int nj = 0; nj < 4; ++nj)
          acc[mi][nj] = wmma_bf16(af2[mi], bf4[nj], acc[mi][nj]);
    }
    __syncthreads();
  }

  // ---- epilogue: bias (+GELU), f32 store (+bf16 shadow) ----
#pragma unroll
  for (int mi = 0; mi < 2; ++mi) {
#pragma unroll
    for (int nj = 0; nj < 4; ++nj) {
      int n = n0 + wn * 64 + nj * 16 + lane15;
      float bv = bias[n];
#pragma unroll
      for (int vj = 0; vj < 8; ++vj) {
        int m = m0 + wm * 32 + mi * 16 + hx * 8 + vj;
        float x = acc[mi][nj][vj] + bv;
        if (GELU_F) x = gelu_f(x);
        C[(size_t)m * N + n] = x;
        if (Cb) Cb[(size_t)m * N + n] = (__bf16)x;
      }
    }
  }
}

// ---------------------------------------------------------------------------
// Banded (sliding-window) attention, flash-style online softmax.
// One wave32 per (qt, chunk, b*H+h); QK^T and P*V via WMMA (16 per key tile);
// only the 17 key tiles inside the +-256 band are visited; k0/v0 global
// column merged analytically. Writes f32 + bf16 shadow (feeds Wo GEMM).
// ---------------------------------------------------------------------------
__global__ __launch_bounds__(32) void band_attn_kernel(
    const float* __restrict__ q, const float* __restrict__ kmat,
    const float* __restrict__ vmat, float* __restrict__ aout,
    __bf16* __restrict__ aout_bf) {
  const int qt = blockIdx.x;          // 0..15
  const int c  = blockIdx.y;          // 0..7
  const int hh = blockIdx.z % NHEAD;
  const int bb = blockIdx.z / NHEAD;
  const int lane = threadIdx.x;
  const int lane15 = lane & 15;
  const int hx = lane >> 4;

  __shared__ __align__(16) __bf16 sQ[32 * 64]; // [m][d], pre-scaled
  __shared__ __align__(16) __bf16 sK[32 * 64]; // [key][d]
  __shared__ __align__(16) __bf16 sV[64 * 32]; // [d][key]
  __shared__ __align__(16) __bf16 sP[32 * 32]; // [m][key]

  for (int idx = lane; idx < 32 * 16; idx += 32) {
    int m = idx >> 4, d4 = (idx & 15) * 4;
    size_t row = (size_t)bb * S_TOT + c * 512 + qt * 32 + m;
    const float4 qv = *(const float4*)(q + row * DM + hh * 64 + d4);
    const float* qf = (const float*)&qv;
#pragma unroll
    for (int j = 0; j < 4; ++j) sQ[m * 64 + d4 + j] = (__bf16)(qf[j] * 0.125f);
  }
  __syncthreads();

  v8f oacc[2][4] = {};
  float mrun[2][8], lrun[2][8];
#pragma unroll
  for (int mi = 0; mi < 2; ++mi)
#pragma unroll
    for (int vj = 0; vj < 8; ++vj) { mrun[mi][vj] = -1e30f; lrun[mi][vj] = 0.f; }

  for (int kt = qt + 8; kt <= qt + 24; ++kt) {
    const int kbase = kt * 32;
    for (int idx = lane; idx < 32 * 16; idx += 32) {
      int key = idx >> 4, d4 = (idx & 15) * 4;
      int jabs = c * 512 + kbase + key - 512;
      float kv[4] = {0.f, 0.f, 0.f, 0.f}, vv[4] = {0.f, 0.f, 0.f, 0.f};
      if (jabs >= 0 && jabs < S_TOT) {
        size_t row = (size_t)bb * S_TOT + jabs;
        const float4 k4 = *(const float4*)(kmat + row * DM + hh * 64 + d4);
        const float4 v4 = *(const float4*)(vmat + row * DM + hh * 64 + d4);
        const float* kp = (const float*)&k4; const float* vp = (const float*)&v4;
#pragma unroll
        for (int j = 0; j < 4; ++j) { kv[j] = kp[j]; vv[j] = vp[j]; }
      }
#pragma unroll
      for (int j = 0; j < 4; ++j) {
        sK[key * 64 + d4 + j]  = (__bf16)kv[j];
        sV[(d4 + j) * 32 + key] = (__bf16)vv[j];
      }
    }
    __syncthreads();

    // ---- S = Q * K^T ----
    v8f sacc[2][2] = {};
#pragma unroll
    for (int ks = 0; ks < 2; ++ks) {
      int kb = ks * 32;
      v16bf aq[2], bk2[2];
#pragma unroll
      for (int mi = 0; mi < 2; ++mi) {
        int m = mi * 16 + lane15;
        aq[mi] = ld_frag(sQ + m * 64 + kb + hx * 8, sQ + m * 64 + kb + 16 + hx * 8);
      }
#pragma unroll
      for (int nj = 0; nj < 2; ++nj) {
        int key = nj * 16 + lane15;
        bk2[nj] = ld_frag(sK + key * 64 + kb + hx * 8, sK + key * 64 + kb + 16 + hx * 8);
      }
#pragma unroll
      for (int mi = 0; mi < 2; ++mi)
#pragma unroll
        for (int nj = 0; nj < 2; ++nj)
          sacc[mi][nj] = wmma_bf16(aq[mi], bk2[nj], sacc[mi][nj]);
    }

    // ---- mask + online softmax, write P (bf16) ----
#pragma unroll
    for (int mi = 0; mi < 2; ++mi) {
      float scv[8];
#pragma unroll
      for (int vj = 0; vj < 8; ++vj) {
        int qi = qt * 32 + mi * 16 + hx * 8 + vj;
        float s0 = sacc[mi][0][vj];
        float s1 = sacc[mi][1][vj];
        int ki0 = kbase + lane15;
        int ki1 = kbase + 16 + lane15;
        int d0 = ki0 - 512 - qi, d1 = ki1 - 512 - qi;
        int j0 = c * 512 + ki0 - 512, j1 = c * 512 + ki1 - 512;
        if (!(d0 >= -256 && d0 <= 256 && j0 >= 1 && j0 < S_TOT)) s0 = -1e9f;
        if (!(d1 >= -256 && d1 <= 256 && j1 >= 1 && j1 < S_TOT)) s1 = -1e9f;
        float tm = rowmax16(fmaxf(s0, s1));
        float mnew = fmaxf(mrun[mi][vj], tm);
        scv[vj] = __expf(mrun[mi][vj] - mnew);
        mrun[mi][vj] = mnew;
        float p0 = (s0 > -1e8f) ? __expf(s0 - mnew) : 0.f;
        float p1 = (s1 > -1e8f) ? __expf(s1 - mnew) : 0.f;
        float rs = rowsum16(p0 + p1);
        lrun[mi][vj] = lrun[mi][vj] * scv[vj] + rs;
        int m = mi * 16 + hx * 8 + vj;
        sP[m * 32 + lane15]      = (__bf16)p0;
        sP[m * 32 + 16 + lane15] = (__bf16)p1;
      }
#pragma unroll
      for (int nd = 0; nd < 4; ++nd)
#pragma unroll
        for (int vj = 0; vj < 8; ++vj)
          oacc[mi][nd][vj] *= scv[vj];
    }
    __syncthreads();

    // ---- O += P * V ----
    v16bf pa[2];
#pragma unroll
    for (int mi = 0; mi < 2; ++mi) {
      int m = mi * 16 + lane15;
      pa[mi] = ld_frag(sP + m * 32 + hx * 8, sP + m * 32 + 16 + hx * 8);
    }
#pragma unroll
    for (int nd = 0; nd < 4; ++nd) {
      int dd = nd * 16 + lane15;
      v16bf vb = ld_frag(sV + dd * 32 + hx * 8, sV + dd * 32 + 16 + hx * 8);
#pragma unroll
      for (int mi = 0; mi < 2; ++mi)
        oacc[mi][nd] = wmma_bf16(pa[mi], vb, oacc[mi][nd]);
    }
    __syncthreads();
  }

  // ---- merge the global-token column (key j=0) and write out ----
  const float* k0p = kmat + (size_t)bb * S_TOT * DM + hh * 64;
  const float* v0p = vmat + (size_t)bb * S_TOT * DM + hh * 64;
  float k0v[4], v0v[4];
#pragma unroll
  for (int j = 0; j < 4; ++j) k0v[j] = k0p[lane15 * 4 + j];
#pragma unroll
  for (int nd = 0; nd < 4; ++nd) v0v[nd] = v0p[nd * 16 + lane15];

#pragma unroll
  for (int mi = 0; mi < 2; ++mi) {
#pragma unroll
    for (int vj = 0; vj < 8; ++vj) {
      int m = mi * 16 + hx * 8 + vj;
      float part = 0.f;
#pragma unroll
      for (int j = 0; j < 4; ++j)
        part += (float)sQ[m * 64 + lane15 * 4 + j] * k0v[j];
      float g = rowsum16(part);
      float mn = fmaxf(mrun[mi][vj], g);
      float scl = __expf(mrun[mi][vj] - mn);
      float p0 = __expf(g - mn);
      float den = lrun[mi][vj] * scl + p0;
      size_t row = (size_t)bb * S_TOT + c * 512 + qt * 32 + m;
#pragma unroll
      for (int nd = 0; nd < 4; ++nd) {
        float o = (oacc[mi][nd][vj] * scl + p0 * v0v[nd]) / den;
        aout[row * DM + hh * 64 + nd * 16 + lane15] = o;
        aout_bf[row * DM + hh * 64 + nd * 16 + lane15] = (__bf16)o;
      }
    }
  }
}

// ---------------------------------------------------------------------------
// Global attention for the CLS token (row 0 of each batch); 24 blocks.
// ---------------------------------------------------------------------------
__global__ __launch_bounds__(256) void global_attn_kernel(
    const float* __restrict__ qg, const float* __restrict__ kg,
    const float* __restrict__ vg, float* __restrict__ aout,
    __bf16* __restrict__ aout_bf) {
  const int bb = blockIdx.x / NHEAD, hh = blockIdx.x % NHEAD;
  const int t = threadIdx.x;
  __shared__ float q0[64];
  __shared__ float sc[S_TOT];
  __shared__ float red[256];
  __shared__ float osum[4][64];
  const size_t base = (size_t)bb * S_TOT * DM + hh * 64;
  if (t < 64) q0[t] = qg[base + t] * 0.125f;
  __syncthreads();
  float lmax = -1e30f;
  for (int s = t; s < S_TOT; s += 256) {
    const float* kr = kg + base + (size_t)s * DM;
    float d = 0.f;
    for (int j = 0; j < 64; ++j) d += q0[j] * kr[j];
    sc[s] = d;
    lmax = fmaxf(lmax, d);
  }
  red[t] = lmax; __syncthreads();
  for (int o = 128; o; o >>= 1) { if (t < o) red[t] = fmaxf(red[t], red[t + o]); __syncthreads(); }
  float mx = red[0]; __syncthreads();
  float ls = 0.f;
  for (int s = t; s < S_TOT; s += 256) { float p = __expf(sc[s] - mx); sc[s] = p; ls += p; }
  red[t] = ls; __syncthreads();
  for (int o = 128; o; o >>= 1) { if (t < o) red[t] += red[t + o]; __syncthreads(); }
  float denom = red[0];
  int d = t & 63, g = t >> 6;
  float acc = 0.f;
  for (int s = g; s < S_TOT; s += 4) acc += sc[s] * vg[base + (size_t)s * DM + d];
  osum[g][d] = acc; __syncthreads();
  if (t < 64) {
    float o = (osum[0][t] + osum[1][t] + osum[2][t] + osum[3][t]) / denom;
    aout[base + t] = o;
    aout_bf[base + t] = (__bf16)o;
  }
}

// ---------------------------------------------------------------------------
// Embedding gather + LayerNorm (one block per token row), f32 + bf16 shadow.
// ---------------------------------------------------------------------------
__global__ __launch_bounds__(256) void embed_ln_kernel(
    const int* __restrict__ ids, const float* __restrict__ we,
    const float* __restrict__ pe, const float* __restrict__ te,
    const float* __restrict__ g, const float* __restrict__ b,
    float* __restrict__ h, __bf16* __restrict__ hb) {
  const int row = blockIdx.x;
  const int s = row & (S_TOT - 1);
  const int t = threadIdx.x;
  const int id = ids[row];
  __shared__ float red[256];
  float x[3]; float ps = 0.f;
#pragma unroll
  for (int i = 0; i < 3; ++i) {
    int cidx = t + i * 256;
    float v = we[(size_t)id * DM + cidx] + pe[(size_t)(s + 2) * DM + cidx] + te[cidx];
    x[i] = v; ps += v;
  }
  red[t] = ps; __syncthreads();
  for (int o = 128; o; o >>= 1) { if (t < o) red[t] += red[t + o]; __syncthreads(); }
  float mean = red[0] * (1.f / DM); __syncthreads();
  float pq = 0.f;
#pragma unroll
  for (int i = 0; i < 3; ++i) { float dd = x[i] - mean; pq += dd * dd; }
  red[t] = pq; __syncthreads();
  for (int o = 128; o; o >>= 1) { if (t < o) red[t] += red[t + o]; __syncthreads(); }
  float rstd = rsqrtf(red[0] * (1.f / DM) + 1e-5f);
#pragma unroll
  for (int i = 0; i < 3; ++i) {
    int cidx = t + i * 256;
    float y = (x[i] - mean) * rstd * g[cidx] + b[cidx];
    h[(size_t)row * DM + cidx] = y;
    hb[(size_t)row * DM + cidx] = (__bf16)y;
  }
}

// y = LN(x + r) * g + b   (one block per row), f32 + bf16 shadow.
__global__ __launch_bounds__(256) void add_ln_kernel(
    const float* __restrict__ xin, const float* __restrict__ r,
    const float* __restrict__ g, const float* __restrict__ b,
    float* __restrict__ out, __bf16* __restrict__ outb) {
  const int row = blockIdx.x;
  const int t = threadIdx.x;
  __shared__ float red[256];
  float x[3]; float ps = 0.f;
#pragma unroll
  for (int i = 0; i < 3; ++i) {
    int cidx = t + i * 256;
    float v = xin[(size_t)row * DM + cidx] + r[(size_t)row * DM + cidx];
    x[i] = v; ps += v;
  }
  red[t] = ps; __syncthreads();
  for (int o = 128; o; o >>= 1) { if (t < o) red[t] += red[t + o]; __syncthreads(); }
  float mean = red[0] * (1.f / DM); __syncthreads();
  float pq = 0.f;
#pragma unroll
  for (int i = 0; i < 3; ++i) { float dd = x[i] - mean; pq += dd * dd; }
  red[t] = pq; __syncthreads();
  for (int o = 128; o; o >>= 1) { if (t < o) red[t] += red[t + o]; __syncthreads(); }
  float rstd = rsqrtf(red[0] * (1.f / DM) + 1e-5f);
#pragma unroll
  for (int i = 0; i < 3; ++i) {
    int cidx = t + i * 256;
    float y = (x[i] - mean) * rstd * g[cidx] + b[cidx];
    out[(size_t)row * DM + cidx] = y;
    outb[(size_t)row * DM + cidx] = (__bf16)y;
  }
}

// out[b] = h[b*S, :] . W_cls + b_cls   (2 blocks)
__global__ __launch_bounds__(256) void cls_kernel(
    const float* __restrict__ h, const float* __restrict__ Wc,
    const float* __restrict__ bc, float* __restrict__ out) {
  const int bb = blockIdx.x;
  const int t = threadIdx.x;
  __shared__ float red[256];
  const float* hr = h + (size_t)bb * S_TOT * DM;
  float ps = 0.f;
#pragma unroll
  for (int i = 0; i < 3; ++i) {
    int cidx = t + i * 256;
    ps += hr[cidx] * Wc[cidx];
  }
  red[t] = ps; __syncthreads();
  for (int o = 128; o; o >>= 1) { if (t < o) red[t] += red[t + o]; __syncthreads(); }
  if (t == 0) out[bb] = red[0] + bc[0];
}

// ---------------------------------------------------------------------------
extern "C" void kernel_launch(void* const* d_in, const int* in_sizes, int n_in,
                              void* d_out, int out_size, void* d_ws, size_t ws_size,
                              hipStream_t stream) {
  (void)in_sizes; (void)n_in; (void)out_size; (void)ws_size;

  const int*   ids   = (const int*)  d_in[0];
  // d_in[1] = mask (unused by reference math)
  const float* wemb  = (const float*)d_in[2];
  const float* pemb  = (const float*)d_in[3];
  const float* temb  = (const float*)d_in[4];
  const float* lneg  = (const float*)d_in[5];
  const float* lneb  = (const float*)d_in[6];
  const float* Wq  = (const float*)d_in[7];   const float* bq  = (const float*)d_in[8];
  const float* Wk  = (const float*)d_in[9];   const float* bk  = (const float*)d_in[10];
  const float* Wv  = (const float*)d_in[11];  const float* bv  = (const float*)d_in[12];
  const float* Wqg = (const float*)d_in[13];  const float* bqg = (const float*)d_in[14];
  const float* Wkg = (const float*)d_in[15];  const float* bkg = (const float*)d_in[16];
  const float* Wvg = (const float*)d_in[17];  const float* bvg = (const float*)d_in[18];
  const float* Wo  = (const float*)d_in[19];  const float* bo  = (const float*)d_in[20];
  const float* ln1g= (const float*)d_in[21];  const float* ln1b= (const float*)d_in[22];
  const float* Wi  = (const float*)d_in[23];  const float* bi  = (const float*)d_in[24];
  const float* Wf  = (const float*)d_in[25];  const float* bff = (const float*)d_in[26];
  const float* ln2g= (const float*)d_in[27];  const float* ln2b= (const float*)d_in[28];
  const float* Wcls= (const float*)d_in[29];  const float* bcls= (const float*)d_in[30];

  const int ROWS = 2 * S_TOT;               // 8192
  const size_t szh = (size_t)ROWS * DM;     // elements per activation tensor
  const size_t szw = (size_t)DM * DM;       // 589824

  // f32 region
  float* h     = (float*)d_ws;
  float* big   = h + szh;                   // 4*szh (q/k/v/spare, FFN mid)
  float* qb    = big;
  float* kb    = big + szh;
  float* vb    = big + 2 * szh;
  float* spare = big + 3 * szh;
  float* abuf  = big + 4 * szh;
  // bf16 region
  __bf16* hbf    = (__bf16*)(abuf + szh);
  __bf16* abufbf = hbf + szh;
  __bf16* bigbf  = abufbf + szh;            // 4*szh bf16 (FFN mid shadow)
  __bf16* wt     = bigbf + 4 * szh;         // weight staging (<= DM*FF elems)
  __bf16* wt0 = wt, *wt1 = wt + szw, *wt2 = wt + 2 * szw;

  const dim3 blk256(256), blk32(32), blkT(32, 8);
  const dim3 gG768(DM / 128, ROWS / 128);      // (6, 64)
  const dim3 gG3072(FFDIM / 128, ROWS / 128);  // (24, 64)
  const dim3 gattn(16, 8, 24);
  const dim3 gT768(DM / 32, DM / 32);          // W[768][768]
  const dim3 gTWi(FFDIM / 32, DM / 32);        // W[768][3072]
  const dim3 gTWf(DM / 32, FFDIM / 32);        // W[3072][768]

  embed_ln_kernel<<<ROWS, blk256, 0, stream>>>(ids, wemb, pemb, temb, lneg, lneb, h, hbf);

  for (int l = 0; l < 12; ++l) {
    const size_t wo   = (size_t)l * szw;
    const size_t boff = (size_t)l * DM;
    const size_t wio  = (size_t)l * DM * FFDIM;
    const size_t bio  = (size_t)l * FFDIM;

    // --- banded self-attention ---
    transpose_bf16_kernel<<<gT768, blkT, 0, stream>>>(Wq + wo, wt0, DM, DM);
    transpose_bf16_kernel<<<gT768, blkT, 0, stream>>>(Wk + wo, wt1, DM, DM);
    transpose_bf16_kernel<<<gT768, blkT, 0, stream>>>(Wv + wo, wt2, DM, DM);
    gemm_kernel<0><<<gG768, blk256, 0, stream>>>(hbf, wt0, bq + boff, qb, (__bf16*)nullptr, ROWS, DM, DM);
    gemm_kernel<0><<<gG768, blk256, 0, stream>>>(hbf, wt1, bk + boff, kb, (__bf16*)nullptr, ROWS, DM, DM);
    gemm_kernel<0><<<gG768, blk256, 0, stream>>>(hbf, wt2, bv + boff, vb, (__bf16*)nullptr, ROWS, DM, DM);
    band_attn_kernel<<<gattn, blk32, 0, stream>>>(qb, kb, vb, abuf, abufbf);

    // --- global (CLS) attention ---
    transpose_bf16_kernel<<<gT768, blkT, 0, stream>>>(Wqg + wo, wt0, DM, DM);
    transpose_bf16_kernel<<<gT768, blkT, 0, stream>>>(Wkg + wo, wt1, DM, DM);
    transpose_bf16_kernel<<<gT768, blkT, 0, stream>>>(Wvg + wo, wt2, DM, DM);
    gemm_kernel<0><<<gG768, blk256, 0, stream>>>(hbf, wt0, bqg + boff, qb, (__bf16*)nullptr, ROWS, DM, DM);
    gemm_kernel<0><<<gG768, blk256, 0, stream>>>(hbf, wt1, bkg + boff, kb, (__bf16*)nullptr, ROWS, DM, DM);
    gemm_kernel<0><<<gG768, blk256, 0, stream>>>(hbf, wt2, bvg + boff, vb, (__bf16*)nullptr, ROWS, DM, DM);
    global_attn_kernel<<<24, blk256, 0, stream>>>(qb, kb, vb, abuf, abufbf);

    // --- output projection + residual LN ---
    transpose_bf16_kernel<<<gT768, blkT, 0, stream>>>(Wo + wo, wt0, DM, DM);
    gemm_kernel<0><<<gG768, blk256, 0, stream>>>(abufbf, wt0, bo + boff, spare, (__bf16*)nullptr, ROWS, DM, DM);
    add_ln_kernel<<<ROWS, blk256, 0, stream>>>(h, spare, ln1g + boff, ln1b + boff, h, hbf);

    // --- FFN ---
    transpose_bf16_kernel<<<gTWi, blkT, 0, stream>>>(Wi + wio, wt, DM, FFDIM);
    gemm_kernel<1><<<gG3072, blk256, 0, stream>>>(hbf, wt, bi + bio, big, bigbf, ROWS, FFDIM, DM);
    transpose_bf16_kernel<<<gTWf, blkT, 0, stream>>>(Wf + wio, wt, FFDIM, DM);
    gemm_kernel<0><<<gG768, blk256, 0, stream>>>(bigbf, wt, bff + boff, abuf, (__bf16*)nullptr, ROWS, DM, FFDIM);
    add_ln_kernel<<<ROWS, blk256, 0, stream>>>(h, abuf, ln2g + boff, ln2b + boff, h, hbf);
  }

  cls_kernel<<<2, blk256, 0, stream>>>(h, Wcls, bcls, (float*)d_out);
}
